// MultiHeadAttention_68152541053005
// MI455X (gfx1250) — compile-verified, matrix operand order checked by reading, not run on hardware
//
#include <hip/hip_runtime.h>
#include <hip/hip_bf16.h>

// ---------------------------------------------------------------------------
// MI455X (gfx1250) multi-head attention forward, bf16-WMMA everywhere.
// B=2, N=2048, D_MODEL=1024, H=16, HEAD_DIM=64.  f32 in / f32 out.
// - every WMMA operand fragment = two ds_load_b128
// - GEMM tiles staged with global_load_b128 / ds_store_b128
// - flash-attention K/V/Q tiles staged with global_load_async_to_lds_b128
//   (ASYNCcnt) via inline asm, synchronized with s_wait_asynccnt
// ---------------------------------------------------------------------------

typedef unsigned int u32;
typedef unsigned long long u64;
typedef __attribute__((ext_vector_type(8)))  float  v8f;
typedef __attribute__((ext_vector_type(16))) __bf16 v16bf;
typedef __attribute__((ext_vector_type(8)))  __bf16 bf16x8;
typedef __attribute__((ext_vector_type(4)))  __bf16 bf16x4;
typedef __attribute__((ext_vector_type(4)))  float  f32x4;

#define B_     2
#define N_     2048
#define DM_    1024
#define H_     16
#define HD_    64
#define M_     (B_ * N_)          // 4096 token rows
#define NQKV_  (3 * DM_)          // 3072 fused qkv columns
#define LOG2E  1.442695041f

// ---------------------------------------------------------------------------
// CDNA5 16-bit fragment gather: per lane (hl = lane>>4) the fragment is two
// contiguous 8-element runs -> two ds_load_b128.
// ---------------------------------------------------------------------------
__device__ __forceinline__ v16bf load_frag(const __bf16* rowbase, int hl) {
    const bf16x8 lo = *(const bf16x8*)(rowbase + 8 * hl);
    const bf16x8 hi = *(const bf16x8*)(rowbase + 16 + 8 * hl);
    return __builtin_shufflevector(lo, hi, 0, 1, 2, 3, 4, 5, 6, 7,
                                   8, 9, 10, 11, 12, 13, 14, 15);
}

// ---------------------------------------------------------------------------
// Async global->LDS 16B copy (CDNA5 GLOBAL_LOAD_ASYNC_TO_LDS_B128, GVS mode).
// vdst = LDS byte offset, vaddr = 32-bit offset VGPR, saddr = 64-bit base.
// Tracked by ASYNCcnt; wait_async() drains it before the workgroup barrier.
// ---------------------------------------------------------------------------
__device__ __forceinline__ void async_load_b128(void* lds_dst, const void* gbase, u32 voff) {
    const u32 ldsoff =
        (u32)(u64)(__attribute__((address_space(3))) char*)lds_dst;
    asm volatile("global_load_async_to_lds_b128 %0, %1, %2"
                 :: "v"(ldsoff), "v"(voff), "s"((u64)gbase)
                 : "memory");
}
__device__ __forceinline__ void wait_async() {
    asm volatile("s_wait_asynccnt 0x0" ::: "memory");
}

// ---------------------------------------------------------------------------
// Stage 0: f32 -> bf16 conversions
// ---------------------------------------------------------------------------
__global__ void cvt_f32_bf16(const float* __restrict__ s, __bf16* __restrict__ d, int n4) {
    for (int i = blockIdx.x * blockDim.x + threadIdx.x; i < n4; i += gridDim.x * blockDim.x)
        ((bf16x4*)d)[i] = __builtin_convertvector(((const f32x4*)s)[i], bf16x4);
}

// Transpose a [rows][cols] f32 weight into bf16 W^T (output row coff+c,
// row length == rows).  Makes GEMM B-fragments contiguous in LDS.
__global__ void cvt_w_transpose(const float* __restrict__ s, __bf16* __restrict__ d,
                                int rows, int cols, int coff) {
    const int total = rows * cols;
    for (int i = blockIdx.x * blockDim.x + threadIdx.x; i < total; i += gridDim.x * blockDim.x) {
        const int r = i / cols, c = i - r * cols;
        d[(coff + c) * rows + r] = (__bf16)s[i];
    }
}

// ---------------------------------------------------------------------------
// Stage 1/4: bf16 WMMA GEMM.  C[M,Ncols] f32 = A[M,K] * B^T[Ncols,K] (bf16).
// block = 256 threads (8 waves).  Block tile 16 x 256, wave tile 16 x 32.
// Tiles staged via global_load_b128 + ds_store_b128.
// ---------------------------------------------------------------------------
__global__ void gemm_bf16_wmma(const __bf16* __restrict__ A, const __bf16* __restrict__ Bt,
                               float* __restrict__ C, int K, int Ncols) {
    __shared__ __align__(16) __bf16 lA[16 * 32];     //  1 KB
    __shared__ __align__(16) __bf16 lBt[256 * 32];   // 16 KB  [col][k]
    const int tid = threadIdx.x;
    const int w = tid >> 5, lane = tid & 31, hl = lane >> 4, ln = lane & 15;
    const int m0 = blockIdx.x * 16, n0 = blockIdx.y * 256;
    const bf16x8* gA8 = (const bf16x8*)A;
    const bf16x8* gB8 = (const bf16x8*)Bt;
    const int K8 = K >> 3;

    v8f acc0, acc1;
#pragma unroll
    for (int r = 0; r < 8; ++r) { acc0[r] = 0.0f; acc1[r] = 0.0f; }

    for (int kk = 0; kk < K; kk += 32) {
        const int kk8 = kk >> 3;
        __syncthreads();
        // stage A tile 16x32 = 64 x 16B (threads 0..63)
        if (tid < 64)
            ((bf16x8*)lA)[tid] = gA8[(m0 + (tid >> 2)) * K8 + kk8 + (tid & 3)];
        // stage B^T tile 256x32 = 1024 x 16B (4 per thread)
#pragma unroll
        for (int i = 0; i < 4; ++i) {
            const int v = tid + i * 256;
            ((bf16x8*)lBt)[v] = gB8[(n0 + (v >> 2)) * K8 + kk8 + (v & 3)];
        }
        if (kk + 32 < K)   // speculative prefetch -> global_prefetch_b8
            __builtin_prefetch(&gB8[(n0 + tid) * K8 + kk8 + 4], 0, 1);
        __syncthreads();

        const v16bf a  = load_frag(&lA[ln * 32], hl);
        const v16bf b0 = load_frag(&lBt[(w * 32 + ln) * 32], hl);
        const v16bf b1 = load_frag(&lBt[(w * 32 + 16 + ln) * 32], hl);
        acc0 = __builtin_amdgcn_wmma_f32_16x16x32_bf16(false, a, false, b0,
                                                       (short)0, acc0, false, false);
        acc1 = __builtin_amdgcn_wmma_f32_16x16x32_bf16(false, a, false, b1,
                                                       (short)0, acc1, false, false);
    }

    const int col0 = n0 + w * 32 + ln;
#pragma unroll
    for (int r = 0; r < 8; ++r) {
        const int row = (m0 + 8 * hl + r) * Ncols;
        C[row + col0]      = acc0[r];
        C[row + col0 + 16] = acc1[r];
    }
}

// ---------------------------------------------------------------------------
// Stage 2: RoPE + per-head RMSNorm + per-dim softplus scale.
// One block per token (512 threads = 16 waves = 16 heads). Lane i owns dims
// {i, i+32} which is exactly the RoPE pair. Writes Q,K as [B][H][N][64] bf16
// and V transposed as [B][H][64][N] bf16 (contiguous flash PV fragments).
// ---------------------------------------------------------------------------
__global__ void rope_norm_kernel(const float* __restrict__ qkv,   // [M][3072]
                                 const float* __restrict__ qln,
                                 const float* __restrict__ kln,
                                 const float* __restrict__ pds,
                                 __bf16* __restrict__ Q, __bf16* __restrict__ Ko,
                                 __bf16* __restrict__ Vt) {
    const int t = blockIdx.x;            // token row, 0..M-1
    const int b = t / N_, n = t - b * N_;
    const int h = threadIdx.x >> 5;      // wave = head
    const int i = threadIdx.x & 31;      // lane = dim in [0,32)

    const float* row = qkv + t * NQKV_;
    const float q1 = row[h * HD_ + i],           q2 = row[h * HD_ + i + 32];
    const float k1 = row[DM_ + h * HD_ + i],     k2 = row[DM_ + h * HD_ + i + 32];
    const float v1 = row[2 * DM_ + h * HD_ + i], v2 = row[2 * DM_ + h * HD_ + i + 32];

    // RoPE: 1/timescale = 10000^(-i/32) = exp2(-i * log2(10000)/32)
    const float inv_ts = exp2f(-(float)i * 0.41524101186f);
    float sn, cs;
    __sincosf((float)n * inv_ts, &sn, &cs);

    const float rq1 = q1 * cs - q2 * sn, rq2 = q2 * cs + q1 * sn;
    const float rk1 = k1 * cs - k2 * sn, rk2 = k2 * cs + k1 * sn;

    // per-head RMSNorm (reduce 64 values across 32 lanes)
    float sq = rq1 * rq1 + rq2 * rq2;
    float sk = rk1 * rk1 + rk2 * rk2;
#pragma unroll
    for (int off = 1; off < 32; off <<= 1) {
        sq += __shfl_xor(sq, off, 32);
        sk += __shfl_xor(sk, off, 32);
    }
    const float rq = rsqrtf(sq * (1.0f / 64.0f) + 1e-6f);
    const float rk = rsqrtf(sk * (1.0f / 64.0f) + 1e-6f);

    // q scale = LOG2_E / sqrt(64) * softplus(per_dim_scale)
    const float s1 = (LOG2E / 8.0f) * log1pf(__expf(pds[i]));
    const float s2 = (LOG2E / 8.0f) * log1pf(__expf(pds[i + 32]));

    const int base  = ((b * H_ + h) * N_ + n) * HD_;
    const int vbase = (b * H_ + h) * HD_ * N_;
    Q[base + i]       = (__bf16)(rq1 * rq * qln[i]      * s1);
    Q[base + i + 32]  = (__bf16)(rq2 * rq * qln[i + 32] * s2);
    Ko[base + i]      = (__bf16)(rk1 * rk * kln[i]);
    Ko[base + i + 32] = (__bf16)(rk2 * rk * kln[i + 32]);
    Vt[vbase + i * N_ + n]        = (__bf16)v1;   // V^T: [dim][n]
    Vt[vbase + (i + 32) * N_ + n] = (__bf16)v2;
}

// ---------------------------------------------------------------------------
// Stage 3: causal flash attention.
// grid = (N/128, H, B); block = 256 (8 waves). Wave w owns q rows
// [q0+16w, q0+16w+16). Q / K / V^T tiles staged with async-to-LDS b128
// (no VGPR round trip); online softmax in WMMA C layout; probs reshaped to
// A layout via a wave-private LDS tile. All fragments are 2x ds_load_b128.
// Writes ctx bf16 as [B*N][H*64] rows for the output projection.
// ---------------------------------------------------------------------------
__global__ void flash_attn_kernel(const __bf16* __restrict__ Q,
                                  const __bf16* __restrict__ K,
                                  const __bf16* __restrict__ Vt,
                                  __bf16* __restrict__ ctx) {
    __shared__ __align__(16) __bf16 lQ[128 * 64];    // 16 KB   [qrow][dim]
    __shared__ __align__(16) __bf16 lK[64 * 64];     //  8 KB   [krow][dim]
    __shared__ __align__(16) __bf16 lVt[64 * 64];    //  8 KB   [dim][krow]
    __shared__ __align__(16) __bf16 lP[8 * 16 * 32]; //  8 KB   per-wave probs

    const int tid = threadIdx.x;
    const int w = tid >> 5, lane = tid & 31, hl = lane >> 4, ln = lane & 15;
    const int q0 = blockIdx.x * 128;
    const int h = blockIdx.y, b = blockIdx.z;
    const int head_base = ((b * H_ + h) * N_) * HD_;   // for Q, K
    const int vt_base   = (b * H_ + h) * HD_ * N_;     // for V^T

    // stage the 128x64 Q tile: 1024 x 16B async copies (4 per thread)
    {
        const void* gQ = (const void*)(Q + head_base + q0 * HD_);
#pragma unroll
        for (int i = 0; i < 4; ++i) {
            const int v = tid + i * 256;
            async_load_b128((char*)lQ + v * 16, gQ, (u32)(v * 16));
        }
        wait_async();
    }
    __syncthreads();

    const v16bf qa0 = load_frag(&lQ[(16 * w + ln) * 64], hl);       // depth 0..31
    const v16bf qa1 = load_frag(&lQ[(16 * w + ln) * 64 + 32], hl);  // depth 32..63

    const int qbase_w = q0 + 16 * w;
    const int wave_qmax = qbase_w + 15;

    float m_i[8], l_i[8];
    v8f o[4];
#pragma unroll
    for (int r = 0; r < 8; ++r) {
        m_i[r] = -3.0e38f; l_i[r] = 0.0f;
        o[0][r] = o[1][r] = o[2][r] = o[3][r] = 0.0f;
    }

    const int nchunks = 2 * blockIdx.x + 2;     // covers k in [0, q0+128)
    for (int c = 0; c < nchunks; ++c) {
        const int kb = c * 64;
        __syncthreads();   // previous chunk fully consumed
        {
            const void* gK  = (const void*)(K + head_base + kb * HD_);
            const void* gVt = (const void*)(Vt + vt_base + kb);
#pragma unroll
            for (int i = 0; i < 2; ++i) {
                const int v = tid + i * 256;            // 512 x 16B per tile
                // K tile: contiguous 8 KB
                async_load_b128((char*)lK + v * 16, gK, (u32)(v * 16));
                // V^T tile: 64 dim-rows x 128B, global row stride N_*2 bytes
                async_load_b128((char*)lVt + v * 16, gVt,
                                (u32)((v >> 3) * (N_ * 2) + (v & 7) * 16));
            }
            wait_async();
        }
        __syncthreads();

#pragma unroll
        for (int sub = 0; sub < 2; ++sub) {
            const int k0 = kb + 32 * sub;
            if (k0 > wave_qmax) break;          // fully above the diagonal

            // ---- scores: 16 q-rows x 32 k-cols, K-depth 64 (B = K^T) ----
            v8f s0, s1;
#pragma unroll
            for (int r = 0; r < 8; ++r) { s0[r] = 0.0f; s1[r] = 0.0f; }
            {
                const __bf16* r0 = &lK[(32 * sub + ln) * 64];        // key row ln
                const __bf16* r1 = &lK[(32 * sub + 16 + ln) * 64];   // key row 16+ln
                s0 = __builtin_amdgcn_wmma_f32_16x16x32_bf16(false, qa0, false,
                        load_frag(r0, hl), (short)0, s0, false, false);
                s0 = __builtin_amdgcn_wmma_f32_16x16x32_bf16(false, qa1, false,
                        load_frag(r0 + 32, hl), (short)0, s0, false, false);
                s1 = __builtin_amdgcn_wmma_f32_16x16x32_bf16(false, qa0, false,
                        load_frag(r1, hl), (short)0, s1, false, false);
                s1 = __builtin_amdgcn_wmma_f32_16x16x32_bf16(false, qa1, false,
                        load_frag(r1 + 32, hl), (short)0, s1, false, false);
            }

            // ---- causal mask + online softmax (rows live in VGPR index) ----
            float mnew[8];
#pragma unroll
            for (int r = 0; r < 8; ++r) {
                const int qrow = qbase_w + 8 * hl + r;
                const float v0 = (k0 + ln      <= qrow) ? s0[r] : -3.0e38f;
                const float v1 = (k0 + 16 + ln <= qrow) ? s1[r] : -3.0e38f;
                s0[r] = v0; s1[r] = v1;
                mnew[r] = fmaxf(v0, v1);
            }
#pragma unroll
            for (int off = 1; off < 16; off <<= 1)
#pragma unroll
                for (int r = 0; r < 8; ++r)
                    mnew[r] = fmaxf(mnew[r], __shfl_xor(mnew[r], off, 32));

            float p0[8], p1[8], rs[8];
#pragma unroll
            for (int r = 0; r < 8; ++r) {
                const float mm = fmaxf(m_i[r], mnew[r]);
                const float sc = __expf(m_i[r] - mm);
                p0[r] = __expf(s0[r] - mm);
                p1[r] = __expf(s1[r] - mm);
                rs[r] = p0[r] + p1[r];
                m_i[r] = mm;
                l_i[r] *= sc;
                o[0][r] *= sc; o[1][r] *= sc; o[2][r] *= sc; o[3][r] *= sc;
            }
#pragma unroll
            for (int off = 1; off < 16; off <<= 1)
#pragma unroll
                for (int r = 0; r < 8; ++r)
                    rs[r] += __shfl_xor(rs[r], off, 32);
#pragma unroll
            for (int r = 0; r < 8; ++r) l_i[r] += rs[r];

            // ---- reshape probs C-layout -> A-layout via wave-private LDS ----
            __bf16* lp = &lP[w * 16 * 32];
#pragma unroll
            for (int r = 0; r < 8; ++r) {
                const int row = 8 * hl + r;
                lp[row * 32 + ln]      = (__bf16)p0[r];
                lp[row * 32 + 16 + ln] = (__bf16)p1[r];
            }
            const v16bf pa = load_frag(&lp[ln * 32], hl);  // DScnt orders RAW

            // ---- PV: probs(16x32) x V(32x64); B from V^T, contiguous ----
#pragma unroll
            for (int d = 0; d < 4; ++d) {
                const __bf16* vr = &lVt[(d * 16 + ln) * 64 + 32 * sub];
                o[d] = __builtin_amdgcn_wmma_f32_16x16x32_bf16(false, pa, false,
                        load_frag(vr, hl), (short)0, o[d], false, false);
            }
        }
    }

    // ---- normalize and write ctx rows [b*N+n][h*64+d] ----
#pragma unroll
    for (int r = 0; r < 8; ++r) {
        const int qrow = qbase_w + 8 * hl + r;
        const float inv = 1.0f / l_i[r];
        __bf16* crow = ctx + (b * N_ + qrow) * DM_ + h * HD_;
#pragma unroll
        for (int d = 0; d < 4; ++d)
            crow[d * 16 + ln] = (__bf16)(o[d][r] * inv);
    }
}

// ---------------------------------------------------------------------------
// launch
// ---------------------------------------------------------------------------
extern "C" void kernel_launch(void* const* d_in, const int* in_sizes, int n_in,
                              void* d_out, int out_size, void* d_ws, size_t ws_size,
                              hipStream_t stream) {
    const float* x   = (const float*)d_in[0];   // [B,N,DM]
    const float* wq  = (const float*)d_in[1];   // [DM,H,HD] == [1024][1024]
    const float* wk  = (const float*)d_in[2];
    const float* wv  = (const float*)d_in[3];
    const float* wo  = (const float*)d_in[4];   // [H,HD,DM] == [1024][1024]
    const float* qln = (const float*)d_in[5];
    const float* kln = (const float*)d_in[6];
    const float* pds = (const float*)d_in[7];
    float* out = (float*)d_out;

    // workspace layout (≈96 MB total)
    char* ws = (char*)d_ws;
    __bf16* Xbf   = (__bf16*)(ws);                 // 4096x1024        (8 MB)
    __bf16* WqkvT = (__bf16*)(ws + 8388608);       // [3072][1024] W^T (6 MB)
    __bf16* WoT   = (__bf16*)(ws + 14680064);      // [1024][1024] W^T (2 MB)
    float*  qkv   = (float*)(ws + 16777216);       // 4096x3072 f32   (48 MB)
    __bf16* Qbf   = (__bf16*)(ws + 67108864);      // [B][H][N][64]    (8 MB)
    __bf16* Kbf   = (__bf16*)(ws + 75497472);      // [B][H][N][64]    (8 MB)
    __bf16* Vtbf  = (__bf16*)(ws + 83886080);      // [B][H][64][N]    (8 MB)
    __bf16* Ctx   = (__bf16*)(ws + 92274688);      // 4096x1024        (8 MB)

    // stage 0: bf16 conversions (+ weight transposes)
    cvt_f32_bf16<<<1024, 256, 0, stream>>>(x, Xbf, (M_ * DM_) / 4);
    cvt_w_transpose<<<2048, 256, 0, stream>>>(wq, WqkvT, DM_, DM_, 0);
    cvt_w_transpose<<<2048, 256, 0, stream>>>(wk, WqkvT, DM_, DM_, DM_);
    cvt_w_transpose<<<2048, 256, 0, stream>>>(wv, WqkvT, DM_, DM_, 2 * DM_);
    cvt_w_transpose<<<2048, 256, 0, stream>>>(wo, WoT, DM_, DM_, 0);

    // stage 1: fused QKV projection  (4096x3072 = X * [Wq|Wk|Wv])
    gemm_bf16_wmma<<<dim3(M_ / 16, NQKV_ / 256), 256, 0, stream>>>(
        Xbf, WqkvT, qkv, DM_, NQKV_);

    // stage 2: RoPE + RMSNorm + per-dim scale -> bf16 heads (V transposed)
    rope_norm_kernel<<<M_, 512, 0, stream>>>(qkv, qln, kln, pds, Qbf, Kbf, Vtbf);

    // stage 3: causal flash attention (async-to-LDS staging)
    flash_attn_kernel<<<dim3(N_ / 128, H_, B_), 256, 0, stream>>>(Qbf, Kbf, Vtbf, Ctx);

    // stage 4: output projection  out = Ctx * Wo
    gemm_bf16_wmma<<<dim3(M_ / 16, DM_ / 256), 256, 0, stream>>>(
        Ctx, WoT, out, DM_, DM_);
}
